// Transformer_Encoder_8400956030978
// MI455X (gfx1250) — matrix operand order repaired and checked by reading, not compiled, not run on hardware
//
#include <hip/hip_runtime.h>
#include <cstdint>
#include <cstddef>

#define DIM       1024
#define HEADS     16
#define DIM_HEAD  64
#define INNER     1024
#define MLP_DIM   4096
#define BATCH     4
#define SEQ       2048
#define TOKENS    (BATCH*SEQ)   // 8192

typedef __attribute__((ext_vector_type(16))) __bf16 v16bf;
typedef __attribute__((ext_vector_type(8)))  __bf16 v8bf;
typedef __attribute__((ext_vector_type(8)))  float  v8f;
typedef __attribute__((ext_vector_type(4)))  int    v4i;

// pointer-to-AS1 v4i (global) and pointer-to-AS3 v4i (LDS)
typedef __attribute__((address_space(1))) v4i* gv4p;
typedef __attribute__((address_space(3))) v4i* lv4p;

union FragU { v16bf f; v8bf h[2]; };

// ---------------------------------------------------------------------------
// Async global->LDS copy of 16 bytes per lane (ASYNCcnt-tracked on gfx1250).
// Falls back to a plain load/ds-store pair if the builtin is unavailable.
// ---------------------------------------------------------------------------
__device__ __forceinline__ void async_copy16(__bf16* lds_dst, const __bf16* gsrc) {
#if __has_builtin(__builtin_amdgcn_global_load_async_to_lds_b128)
  __builtin_amdgcn_global_load_async_to_lds_b128(
      (gv4p)const_cast<__bf16*>(gsrc), (lv4p)lds_dst, 0, 0);
#else
  *(v8bf*)lds_dst = *(const v8bf*)gsrc;
#endif
}

__device__ __forceinline__ void wait_async() {
#if __has_builtin(__builtin_amdgcn_s_wait_asynccnt)
  __builtin_amdgcn_s_wait_asynccnt(0);
#elif defined(__gfx1250__)
  asm volatile("s_wait_asynccnt 0" ::: "memory");
#endif
}

// ---------------------------------------------------------------------------
// WMMA fragment loads (bf16, 16x16x32).
// A (16xK=32, MxK): lane<16 -> row=lane, K 0..7 & 16..23; lane>=16 -> K 8..15 & 24..31
// B (K=32x16, KxN) read from Bt (NxK row-major): lane<16 -> col n, K 0..7 & 16..23; lane>=16 -> K 8..15 & 24..31
// ---------------------------------------------------------------------------
__device__ __forceinline__ v16bf frag_ld_a(const __bf16* base, int stride) {
  unsigned lane = threadIdx.x & 31u;
  unsigned row  = lane & 15u;
  unsigned kh   = (lane >> 4) << 3;               // 0 or 8
  const __bf16* p = base + (size_t)row * stride + kh;
  FragU r;
  r.h[0] = *(const v8bf*)(p);                     // K = kh .. kh+7
  r.h[1] = *(const v8bf*)(p + 16);                // K = kh+16 .. kh+23
  return r.f;
}

__device__ __forceinline__ v16bf frag_ld_b(const __bf16* base, int stride) {
  unsigned lane = threadIdx.x & 31u;
  unsigned col  = lane & 15u;
  unsigned kh   = (lane >> 4) << 3;               // 0 or 8
  const __bf16* p = base + (size_t)col * stride + kh;
  FragU r;
  r.h[0] = *(const v8bf*)(p);                     // K = kh .. kh+7
  r.h[1] = *(const v8bf*)(p + 16);                // K = kh+16 .. kh+23
  return r.f;
}

__device__ __forceinline__ v8f wmma_bf16(v16bf a, v16bf b, v8f c) {
  // D = A*B + C  (f32 accumulate)
  return __builtin_amdgcn_wmma_f32_16x16x32_bf16(false, a, false, b, (short)0, c,
                                                 false, false);
}

__device__ __forceinline__ float gelu_f(float x) {
  const float c = 1.2533141373155003f;            // sqrt(pi/2), faithful to source
  float x3 = x * x * x;
  return 0.5f * x * (1.0f + tanhf(c * (x + 0.044715f * x3)));
}

// ---------------------------------------------------------------------------
// Weight transpose + fp32 -> bf16:  W[K][N] -> Wt[N][K]
// ---------------------------------------------------------------------------
__global__ __launch_bounds__(256) void transp_to_bf16(const float* __restrict__ W,
                                                      __bf16* __restrict__ Wt,
                                                      int K, int N) {
  long long i = (long long)blockIdx.x * 256 + threadIdx.x;
  if (i < (long long)K * N) {
    int n = (int)(i / K);
    int k = (int)(i % K);
    Wt[i] = (__bf16)W[(size_t)k * N + n];
  }
}

// ---------------------------------------------------------------------------
// LayerNorm (row = DIM=1024 f32) -> bf16
// ---------------------------------------------------------------------------
__global__ __launch_bounds__(256) void ln_to_bf16(const float* __restrict__ x,
                                                  const float* __restrict__ g,
                                                  const float* __restrict__ bt,
                                                  __bf16* __restrict__ out) {
  int row = blockIdx.x;
  const float* xr = x + (size_t)row * DIM;
  float vals[4];
  float s = 0.f, s2 = 0.f;
#pragma unroll
  for (int i = 0; i < 4; ++i) {
    float v = xr[threadIdx.x + i * 256];
    vals[i] = v; s += v; s2 += v * v;
  }
#pragma unroll
  for (int off = 16; off >= 1; off >>= 1) {
    s  += __shfl_xor(s,  off, 32);
    s2 += __shfl_xor(s2, off, 32);
  }
  __shared__ float rs[8], rs2[8];
  unsigned wid = threadIdx.x >> 5;
  if ((threadIdx.x & 31u) == 0) { rs[wid] = s; rs2[wid] = s2; }
  __syncthreads();
  s = 0.f; s2 = 0.f;
#pragma unroll
  for (int w = 0; w < 8; ++w) { s += rs[w]; s2 += rs2[w]; }
  float mean = s * (1.0f / DIM);
  float var  = s2 * (1.0f / DIM) - mean * mean;
  float inv  = rsqrtf(var + 1e-5f);
#pragma unroll
  for (int i = 0; i < 4; ++i) {
    int c = threadIdx.x + i * 256;
    out[(size_t)row * DIM + c] = (__bf16)((vals[i] - mean) * inv * g[c] + bt[c]);
  }
}

// ---------------------------------------------------------------------------
// Tiled WMMA GEMM:  C[M][N] = A[M][K] * Wt[N][K]^T  (+ epilogue)
// Block tile 128x128, BLOCK_K=32, 8 waves: wave_m = wid&3 (32 rows),
// wave_n = wid>>2 (64 cols) -> 2x4 accumulators per wave.
// EPI: 0 = bf16 raw, 1 = bf16 gelu(c+bias), 2 = f32 c+bias+resid
// ---------------------------------------------------------------------------
#define LDK 40   // padded LDS K-stride (elements)

__device__ __forceinline__ void copy_tile(__bf16* __restrict__ dst,
                                          const __bf16* __restrict__ src, int ldk) {
#pragma unroll
  for (int i = 0; i < 2; ++i) {
    unsigned c   = threadIdx.x + (unsigned)i * 256u;
    unsigned row = c >> 2;
    unsigned col = (c & 3u) << 3;
    async_copy16(dst + row * LDK + col, src + (size_t)row * ldk + col);
  }
}

__device__ __forceinline__ void prefetch_tile(const __bf16* src, int ldk) {
#pragma unroll
  for (int i = 0; i < 2; ++i) {
    unsigned c   = threadIdx.x + (unsigned)i * 256u;
    unsigned row = c >> 2;
    unsigned col = (c & 3u) << 3;
    __builtin_prefetch((const void*)(src + (size_t)row * ldk + col), 0, 1);
  }
}

template <int EPI>
__global__ __launch_bounds__(256) void gemm_bf16(const __bf16* __restrict__ A,
                                                 const __bf16* __restrict__ Bt,
                                                 int M, int Nn, int K,
                                                 const float* __restrict__ bias,
                                                 const float* __restrict__ resid,
                                                 void* __restrict__ Cout) {
  __shared__ __bf16 sA[2][128 * LDK];
  __shared__ __bf16 sB[2][128 * LDK];

  int bm = blockIdx.y * 128;
  int bn = blockIdx.x * 128;
  unsigned wid = threadIdx.x >> 5;
  unsigned wm  = wid & 3u;     // 0..3 : 32 rows each
  unsigned wn  = wid >> 2;     // 0..1 : 64 cols each
  unsigned lane  = threadIdx.x & 31u;
  unsigned rbase = (lane >> 4) << 3;   // 0 or 8
  unsigned cn    = lane & 15u;

  v8f acc[2][4] = {};

  const __bf16* Ab = A  + (size_t)bm * K;
  const __bf16* Bb = Bt + (size_t)bn * K;

  copy_tile(sA[0], Ab, K);
  copy_tile(sB[0], Bb, K);
  wait_async();
  __syncthreads();

  int nk = K >> 5;
  for (int kk = 0; kk < nk; ++kk) {
    int cur = kk & 1, nxt = cur ^ 1;
    if (kk + 1 < nk) {
      copy_tile(sA[nxt], Ab + (size_t)(kk + 1) * 32, K);
      copy_tile(sB[nxt], Bb + (size_t)(kk + 1) * 32, K);
      if (kk + 2 < nk) {
        prefetch_tile(Ab + (size_t)(kk + 2) * 32, K);
        prefetch_tile(Bb + (size_t)(kk + 2) * 32, K);
      }
    }
    v16bf af[2], bfv[4];
#pragma unroll
    for (int mt = 0; mt < 2; ++mt)
      af[mt] = frag_ld_a(sA[cur] + (wm * 32 + mt * 16) * LDK, LDK);
#pragma unroll
    for (int nt = 0; nt < 4; ++nt)
      bfv[nt] = frag_ld_b(sB[cur] + (wn * 64 + nt * 16) * LDK, LDK);
#pragma unroll
    for (int mt = 0; mt < 2; ++mt)
#pragma unroll
      for (int nt = 0; nt < 4; ++nt)
        acc[mt][nt] = wmma_bf16(af[mt], bfv[nt], acc[mt][nt]);
    wait_async();
    __syncthreads();
  }

  // epilogue
#pragma unroll
  for (int mt = 0; mt < 2; ++mt) {
#pragma unroll
    for (int nt = 0; nt < 4; ++nt) {
      int n = bn + (int)(wn * 64 + nt * 16 + cn);
#pragma unroll
      for (int r = 0; r < 8; ++r) {
        int m = bm + (int)(wm * 32 + mt * 16 + rbase + r);
        float val = acc[mt][nt][r];
        if (EPI == 1) val = gelu_f(val + bias[n]);
        if (EPI == 2) val = val + bias[n] + resid[(size_t)m * Nn + n];
        if (EPI == 2) ((float*)Cout)[(size_t)m * Nn + n] = val;
        else          ((__bf16*)Cout)[(size_t)m * Nn + n] = (__bf16)val;
      }
    }
  }
}

// ---------------------------------------------------------------------------
// Flash attention. grid = (SEQ/128, HEADS, BATCH), block = 256 (8 waves).
// Each wave owns 16 query rows; key blocks of 64 streamed through LDS.
// scale = DIM^-0.5 = 1/32 (reference scales by dim, not dim_head).
// ---------------------------------------------------------------------------
#define AKS 72   // padded LDS stride for 64-wide tiles

__global__ __launch_bounds__(256) void attn_kernel(const __bf16* __restrict__ q,
                                                   const __bf16* __restrict__ k,
                                                   const __bf16* __restrict__ v,
                                                   const int* __restrict__ mask,
                                                   __bf16* __restrict__ out) {
  __shared__ __bf16 sK[64 * AKS];        // [key][d]
  __shared__ __bf16 sVt[64 * AKS];       // [d][key]
  __shared__ __bf16 sP[8 * 16 * AKS];    // per-wave P tile [16][keys=64]

  int b  = blockIdx.z;
  int h  = blockIdx.y;
  int qb = blockIdx.x * 128;
  unsigned tid   = threadIdx.x;
  unsigned wid   = tid >> 5;
  unsigned lane  = tid & 31u;
  unsigned rbase = (lane >> 4) << 3;     // 0 or 8
  unsigned cn    = lane & 15u;

  const __bf16* qw = q + ((size_t)(b * SEQ + qb + (int)wid * 16)) * INNER + h * DIM_HEAD;
  v16bf qf[2];
  qf[0] = frag_ld_a(qw, INNER);
  qf[1] = frag_ld_a(qw + 32, INNER);

  const int* mrow = mask + (size_t)b * SEQ;   // mask is (B,1,1,N)

  float rmax[8], rsum[8];
  v8f o[4] = {};
#pragma unroll
  for (int r = 0; r < 8; ++r) { rmax[r] = -1e30f; rsum[r] = 0.f; }

  const float scale = 0.03125f;          // 1024^-0.5

  for (int kb = 0; kb < SEQ; kb += 64) {
    __syncthreads();   // previous iteration done with sK/sVt
    // --- stage K block [64][64] row-major (async global->LDS) ---
    const __bf16* kp = k + ((size_t)(b * SEQ + kb)) * INNER + h * DIM_HEAD;
#pragma unroll
    for (int i = 0; i < 2; ++i) {
      unsigned c = tid + (unsigned)i * 256u;
      unsigned row = c >> 3, col = (c & 7u) << 3;
      async_copy16(sK + row * AKS + col, kp + (size_t)row * INNER + col);
    }
    // --- stage V transposed: sVt[d][key] ---
    const __bf16* vp = v + ((size_t)(b * SEQ + kb)) * INNER + h * DIM_HEAD;
    for (int i = tid; i < 64 * 64; i += 256) {
      int key = i >> 6, d = i & 63;
      sVt[d * AKS + key] = vp[(size_t)key * INNER + d];
    }
    wait_async();
    __syncthreads();

    // --- S = Q * K^T  (16 rows x 64 keys per wave) ---
    v8f s[4];
#pragma unroll
    for (int nt = 0; nt < 4; ++nt) {
      v8f a = {};
      v16bf b0 = frag_ld_b(sK + (nt * 16) * AKS + 0, AKS);
      v16bf b1 = frag_ld_b(sK + (nt * 16) * AKS + 32, AKS);
      a = wmma_bf16(qf[0], b0, a);
      a = wmma_bf16(qf[1], b1, a);
      s[nt] = a;
    }
    // scale + mask
#pragma unroll
    for (int nt = 0; nt < 4; ++nt) {
      int mk = mrow[kb + nt * 16 + (int)cn];
#pragma unroll
      for (int r = 0; r < 8; ++r) {
        float t = s[nt][r] * scale;
        if (mk == 0) t = -65500.0f;
        s[nt][r] = t;
      }
    }
    // --- online softmax (per-row over 4 tiles x 16 lanes of half-wave) ---
    float mx[8];
#pragma unroll
    for (int r = 0; r < 8; ++r) {
      float m0 = fmaxf(fmaxf(s[0][r], s[1][r]), fmaxf(s[2][r], s[3][r]));
      mx[r] = m0;
    }
#pragma unroll
    for (int off = 8; off >= 1; off >>= 1)
#pragma unroll
      for (int r = 0; r < 8; ++r)
        mx[r] = fmaxf(mx[r], __shfl_xor(mx[r], off, 32));

    float corr[8];
#pragma unroll
    for (int r = 0; r < 8; ++r) {
      float mn = fmaxf(rmax[r], mx[r]);
      corr[r] = __expf(rmax[r] - mn);
      rmax[r] = mn;
    }
    float psum[8];
#pragma unroll
    for (int r = 0; r < 8; ++r) psum[r] = 0.f;
#pragma unroll
    for (int nt = 0; nt < 4; ++nt)
#pragma unroll
      for (int r = 0; r < 8; ++r) {
        float p = __expf(s[nt][r] - rmax[r]);
        s[nt][r] = p;
        psum[r] += p;
      }
#pragma unroll
    for (int off = 8; off >= 1; off >>= 1)
#pragma unroll
      for (int r = 0; r < 8; ++r)
        psum[r] += __shfl_xor(psum[r], off, 32);
#pragma unroll
    for (int r = 0; r < 8; ++r) rsum[r] = rsum[r] * corr[r] + psum[r];
#pragma unroll
    for (int ot = 0; ot < 4; ++ot)
#pragma unroll
      for (int r = 0; r < 8; ++r)
        o[ot][r] *= corr[r];

    // --- P -> LDS (per-wave region, transposes C-layout into A-layout) ---
    __bf16* myP = sP + (size_t)wid * 16 * AKS;
#pragma unroll
    for (int nt = 0; nt < 4; ++nt)
#pragma unroll
      for (int r = 0; r < 8; ++r)
        myP[(rbase + r) * AKS + nt * 16 + cn] = (__bf16)s[nt][r];

    v16bf pf[2];
    pf[0] = frag_ld_a(myP, AKS);
    pf[1] = frag_ld_a(myP + 32, AKS);

    // --- O += P * V ---
#pragma unroll
    for (int ot = 0; ot < 4; ++ot) {
      v16bf vf0 = frag_ld_b(sVt + (ot * 16) * AKS + 0, AKS);
      v16bf vf1 = frag_ld_b(sVt + (ot * 16) * AKS + 32, AKS);
      o[ot] = wmma_bf16(pf[0], vf0, o[ot]);
      o[ot] = wmma_bf16(pf[1], vf1, o[ot]);
    }
  }

  // --- normalize + store bf16 ---
#pragma unroll
  for (int ot = 0; ot < 4; ++ot)
#pragma unroll
    for (int r = 0; r < 8; ++r) {
      int m = qb + (int)(wid * 16 + rbase + r);
      int d = ot * 16 + (int)cn;
      out[((size_t)(b * SEQ + m)) * INNER + h * DIM_HEAD + d] =
          (__bf16)(o[ot][r] / rsum[r]);
    }
}

// ---------------------------------------------------------------------------
// Orchestration
// ---------------------------------------------------------------------------
extern "C" void kernel_launch(void* const* d_in, const int* in_sizes, int n_in,
                              void* d_out, int out_size, void* d_ws, size_t ws_size,
                              hipStream_t stream) {
  const float* x     = (const float*)d_in[0];
  const int*   mask  = (const int*)  d_in[1];
  const float* wq    = (const float*)d_in[2];
  const float* wk    = (const float*)d_in[3];
  const float* wv    = (const float*)d_in[4];
  const float* wo    = (const float*)d_in[5];
  const float* bo    = (const float*)d_in[6];
  const float* w1    = (const float*)d_in[7];
  const float* b1    = (const float*)d_in[8];
  const float* w2    = (const float*)d_in[9];
  const float* b2    = (const float*)d_in[10];
  const float* ln1g  = (const float*)d_in[11];
  const float* ln1b  = (const float*)d_in[12];
  const float* ln2g  = (const float*)d_in[13];
  const float* ln2b  = (const float*)d_in[14];

  const size_t MB = 1ull << 20;
  char* ws = (char*)d_ws;
  __bf16* wt_q = (__bf16*)(ws + 0 * MB);     // [1024][1024]
  __bf16* wt_k = (__bf16*)(ws + 2 * MB);
  __bf16* wt_v = (__bf16*)(ws + 4 * MB);
  __bf16* wt_o = (__bf16*)(ws + 6 * MB);
  __bf16* w1t  = (__bf16*)(ws + 8 * MB);     // [4096][1024]
  __bf16* w2t  = (__bf16*)(ws + 16 * MB);    // [1024][4096]
  __bf16* h_bf = (__bf16*)(ws + 24 * MB);    // [8192][1024]   (ln1, later ln2)
  __bf16* qb   = (__bf16*)(ws + 40 * MB);    // [8192][1024]
  __bf16* kb   = (__bf16*)(ws + 56 * MB);
  __bf16* vb   = (__bf16*)(ws + 72 * MB);
  __bf16* ao   = (__bf16*)(ws + 88 * MB);    // attn out
  __bf16* ff   = qb;                         // [8192][4096] reuses q/k/v/ao
  float*  x1   = (float*)(ws + 104 * MB);    // [8192][1024] f32 residual

  // 1) weight transpose/convert
  transp_to_bf16<<<(DIM * INNER + 255) / 256, 256, 0, stream>>>(wq, wt_q, DIM, INNER);
  transp_to_bf16<<<(DIM * INNER + 255) / 256, 256, 0, stream>>>(wk, wt_k, DIM, INNER);
  transp_to_bf16<<<(DIM * INNER + 255) / 256, 256, 0, stream>>>(wv, wt_v, DIM, INNER);
  transp_to_bf16<<<(INNER * DIM + 255) / 256, 256, 0, stream>>>(wo, wt_o, INNER, DIM);
  transp_to_bf16<<<(DIM * MLP_DIM + 255) / 256, 256, 0, stream>>>(w1, w1t, DIM, MLP_DIM);
  transp_to_bf16<<<(MLP_DIM * DIM + 255) / 256, 256, 0, stream>>>(w2, w2t, MLP_DIM, DIM);

  // 2) LN1
  ln_to_bf16<<<TOKENS, 256, 0, stream>>>(x, ln1g, ln1b, h_bf);

  // 3) QKV projections
  dim3 gqkv(INNER / 128, TOKENS / 128);
  gemm_bf16<0><<<gqkv, 256, 0, stream>>>(h_bf, wt_q, TOKENS, INNER, DIM, nullptr, nullptr, qb);
  gemm_bf16<0><<<gqkv, 256, 0, stream>>>(h_bf, wt_k, TOKENS, INNER, DIM, nullptr, nullptr, kb);
  gemm_bf16<0><<<gqkv, 256, 0, stream>>>(h_bf, wt_v, TOKENS, INNER, DIM, nullptr, nullptr, vb);

  // 4) attention
  attn_kernel<<<dim3(SEQ / 128, HEADS, BATCH), 256, 0, stream>>>(qb, kb, vb, mask, ao);

  // 5) O-proj + bias + residual -> x1 (f32)
  gemm_bf16<2><<<dim3(DIM / 128, TOKENS / 128), 256, 0, stream>>>(
      ao, wt_o, TOKENS, DIM, INNER, bo, x, x1);

  // 6) LN2
  ln_to_bf16<<<TOKENS, 256, 0, stream>>>(x1, ln2g, ln2b, h_bf);

  // 7) MLP up + GELU
  gemm_bf16<1><<<dim3(MLP_DIM / 128, TOKENS / 128), 256, 0, stream>>>(
      h_bf, w1t, TOKENS, MLP_DIM, DIM, b1, nullptr, ff);

  // 8) MLP down + bias + residual -> out (f32)
  gemm_bf16<2><<<dim3(DIM / 128, TOKENS / 128), 256, 0, stream>>>(
      ff, w2t, TOKENS, DIM, MLP_DIM, b2, x1, (float*)d_out);
}
